// NeuralODE_19078244729125
// MI455X (gfx1250) — compile-verified, matrix operand order checked by reading, not run on hardware
//
#include <hip/hip_runtime.h>
#include <hip/hip_bf16.h>

typedef __attribute__((ext_vector_type(16))) _Float16 v16h;
typedef __attribute__((ext_vector_type(2)))  _Float16 v2h;
typedef __attribute__((ext_vector_type(2)))  __fp16   v2hf;
typedef __attribute__((ext_vector_type(8)))  float    v8f;

// ---- fast tanh: gfx1250 native V_TANH_F32 (confirmed in disasm) ----
#if __has_builtin(__builtin_amdgcn_tanhf)
#define FAST_TANH(x) __builtin_amdgcn_tanhf(x)
#elif __has_builtin(__builtin_amdgcn_tanh_f32)
#define FAST_TANH(x) __builtin_amdgcn_tanh_f32(x)
#else
__device__ __forceinline__ float fast_tanh_impl(float x) {
  float ax = __builtin_fabsf(x);
  float e = __builtin_amdgcn_exp2f(ax * -2.8853900817779268f);
  float r = (1.0f - e) * __builtin_amdgcn_rcpf(1.0f + e);
  return __builtin_copysignf(r, x);
}
#define FAST_TANH(x) fast_tanh_impl(x)
#endif

__device__ __forceinline__ v2h pk2(float a, float b) {
#if __has_builtin(__builtin_amdgcn_cvt_pkrtz)
  v2hf t = __builtin_amdgcn_cvt_pkrtz(a, b);   // v_cvt_pk_rtz_f16_f32
  return __builtin_bit_cast(v2h, t);
#else
  v2h p; p[0] = (_Float16)a; p[1] = (_Float16)b; return p;
#endif
}

#define WMMA_F16(A, Bm, C) \
  __builtin_amdgcn_wmma_f32_16x16x32_f16(false, (A), false, (Bm), (short)0, (C), false, false)

__global__ __launch_bounds__(256) void node_dopri5_wmma(
    const float* __restrict__ y0p, const float* __restrict__ tp,
    const float* __restrict__ W1p, const float* __restrict__ b1p,
    const float* __restrict__ W2p, const float* __restrict__ b2p,
    float* __restrict__ outp, int B, int N) {
  const int tid  = blockIdx.x * blockDim.x + threadIdx.x;
  const int lane = threadIdx.x & 31;
  const int wave = tid >> 5;
  const int bb   = wave * 32;           // two independent 16-batch WMMA tiles per wave
  if (bb >= B) return;
  const int  m      = lane & 15;
  const bool hiHalf = lane >= 16;

  // ---- preload weight tiles into WMMA operand layouts (one time) ----
  // f16 A-layout (16x32, M across lanes): elem e of lane holds K = e + (e>=8?8:0) + (hi?8:0)
  v16h A1lo = {}, A1hi = {}, A2 = {};
#pragma unroll
  for (int e = 0; e < 16; ++e) {
    const int K = e + (e >= 8 ? 8 : 0) + (hiHalf ? 8 : 0);
    A1lo[e] = (K < 2) ? (_Float16)W1p[K * 32 + m]      : (_Float16)0.0f;  // W1^T, hidden 0..15
    A1hi[e] = (K < 2) ? (_Float16)W1p[K * 32 + 16 + m] : (_Float16)0.0f;  // hidden 16..31
    A2[e]   = (m < 2) ? (_Float16)W2p[K * 2 + m]       : (_Float16)0.0f;  // W2^T (2x32)
  }
  // f32 C-layout bias tiles: VGPR j holds row j (lanes 0-15) / row j+8 (lanes 16-31)
  const int r0 = hiHalf ? 8 : 0;
  v8f C1lo, C1hi, C2;
#pragma unroll
  for (int j = 0; j < 8; ++j) {
    C1lo[j] = b1p[r0 + j];
    C1hi[j] = b1p[16 + r0 + j];
    C2[j]   = (r0 + j) < 2 ? b2p[r0 + j] : 0.0f;
  }

  // ---- two streams: s=0 -> batch bb..bb+15, s=1 -> bb+16..bb+31 ----
  // Persistent B1 operands: elements 2..15 stay zero; only first VGPR rewritten per f-eval.
  v16h B1a = {}, B1b = {};
  float yx0 = 0.0f, yy0 = 0.0f, yx1 = 0.0f, yy1 = 0.0f;
  const bool act0 = (!hiHalf) && (bb + m < B);
  const bool act1 = (!hiHalf) && (bb + 16 + m < B);
  if (act0) { yx0 = y0p[(size_t)(bb + m) * 2];      yy0 = y0p[(size_t)(bb + m) * 2 + 1]; }
  if (act1) { yx1 = y0p[(size_t)(bb + 16 + m) * 2]; yy1 = y0p[(size_t)(bb + 16 + m) * 2 + 1]; }

  // Evaluate f on BOTH tiles, explicitly interleaved so each WMMA has >=3 other
  // WMMAs plus independent tanh/pack VALU between it and its consumer (kills
  // the 4-op WMMA->VALU hazard nops).
  auto feval2 = [&](float ax0, float ay0, float ax1, float ay1,
                    float& ox0, float& oy0, float& ox1, float& oy1) {
    v2h p0 = pk2(ax0, ay0);
    v2h p1 = pk2(ax1, ay1);
    B1a[0] = p0[0]; B1a[1] = p0[1];
    B1b[0] = p1[0]; B1b[1] = p1[1];
    // four independent layer-1 WMMAs back-to-back
    v8f w0 = WMMA_F16(A1lo, B1a, C1lo);
    v8f w1 = WMMA_F16(A1hi, B1a, C1hi);
    v8f w2 = WMMA_F16(A1lo, B1b, C1lo);
    v8f w3 = WMMA_F16(A1hi, B1b, C1hi);
    // tanh + pack stream 0 while stream 1 WMMAs drain
    v16h B2a, B2b;
#pragma unroll
    for (int j = 0; j < 8; ++j) w0[j] = FAST_TANH(w0[j]);
#pragma unroll
    for (int j = 0; j < 8; ++j) w1[j] = FAST_TANH(w1[j]);
#pragma unroll
    for (int v = 0; v < 4; ++v) {
      v2h a = pk2(w0[2 * v], w0[2 * v + 1]);
      v2h b = pk2(w1[2 * v], w1[2 * v + 1]);
      B2a[2 * v] = a[0]; B2a[2 * v + 1] = a[1];
      B2a[8 + 2 * v] = b[0]; B2a[9 + 2 * v] = b[1];
    }
#pragma unroll
    for (int j = 0; j < 8; ++j) w2[j] = FAST_TANH(w2[j]);
#pragma unroll
    for (int j = 0; j < 8; ++j) w3[j] = FAST_TANH(w3[j]);
#pragma unroll
    for (int v = 0; v < 4; ++v) {
      v2h a = pk2(w2[2 * v], w2[2 * v + 1]);
      v2h b = pk2(w3[2 * v], w3[2 * v + 1]);
      B2b[2 * v] = a[0]; B2b[2 * v + 1] = a[1];
      B2b[8 + 2 * v] = b[0]; B2b[9 + 2 * v] = b[1];
    }
    // two independent layer-2 WMMAs back-to-back
    v8f d0 = WMMA_F16(A2, B2a, C2);
    v8f d1 = WMMA_F16(A2, B2b, C2);
    ox0 = d0[0]; oy0 = d0[1];           // rows M=0,1 live in lanes 0..15
    ox1 = d1[0]; oy1 = d1[1];
  };

  float2* outv = reinterpret_cast<float2*>(outp);
  if (act0) outv[(size_t)0 * B + bb + m]      = make_float2(yx0, yy0);
  if (act1) outv[(size_t)0 * B + bb + 16 + m] = make_float2(yx1, yy1);

  for (int i = 0; i < N - 1; ++i) {
    const float h = tp[i + 1] - tp[i];  // uniform scalar loads (t cached in K$)
    float k1x0, k1y0, k1x1, k1y1, k2x0, k2y0, k2x1, k2y1;
    float k3x0, k3y0, k3x1, k3y1, k4x0, k4y0, k4x1, k4y1;
    float k5x0, k5y0, k5x1, k5y1, k6x0, k6y0, k6x1, k6y1;

    feval2(yx0, yy0, yx1, yy1, k1x0, k1y0, k1x1, k1y1);

    feval2(yx0 + h * (0.2f * k1x0), yy0 + h * (0.2f * k1y0),
           yx1 + h * (0.2f * k1x1), yy1 + h * (0.2f * k1y1),
           k2x0, k2y0, k2x1, k2y1);

    feval2(yx0 + h * ((3.0f / 40.0f) * k1x0 + (9.0f / 40.0f) * k2x0),
           yy0 + h * ((3.0f / 40.0f) * k1y0 + (9.0f / 40.0f) * k2y0),
           yx1 + h * ((3.0f / 40.0f) * k1x1 + (9.0f / 40.0f) * k2x1),
           yy1 + h * ((3.0f / 40.0f) * k1y1 + (9.0f / 40.0f) * k2y1),
           k3x0, k3y0, k3x1, k3y1);

    feval2(yx0 + h * ((44.0f / 45.0f) * k1x0 - (56.0f / 15.0f) * k2x0 + (32.0f / 9.0f) * k3x0),
           yy0 + h * ((44.0f / 45.0f) * k1y0 - (56.0f / 15.0f) * k2y0 + (32.0f / 9.0f) * k3y0),
           yx1 + h * ((44.0f / 45.0f) * k1x1 - (56.0f / 15.0f) * k2x1 + (32.0f / 9.0f) * k3x1),
           yy1 + h * ((44.0f / 45.0f) * k1y1 - (56.0f / 15.0f) * k2y1 + (32.0f / 9.0f) * k3y1),
           k4x0, k4y0, k4x1, k4y1);

    feval2(yx0 + h * ((19372.0f / 6561.0f) * k1x0 - (25360.0f / 2187.0f) * k2x0
                      + (64448.0f / 6561.0f) * k3x0 - (212.0f / 729.0f) * k4x0),
           yy0 + h * ((19372.0f / 6561.0f) * k1y0 - (25360.0f / 2187.0f) * k2y0
                      + (64448.0f / 6561.0f) * k3y0 - (212.0f / 729.0f) * k4y0),
           yx1 + h * ((19372.0f / 6561.0f) * k1x1 - (25360.0f / 2187.0f) * k2x1
                      + (64448.0f / 6561.0f) * k3x1 - (212.0f / 729.0f) * k4x1),
           yy1 + h * ((19372.0f / 6561.0f) * k1y1 - (25360.0f / 2187.0f) * k2y1
                      + (64448.0f / 6561.0f) * k3y1 - (212.0f / 729.0f) * k4y1),
           k5x0, k5y0, k5x1, k5y1);

    feval2(yx0 + h * ((9017.0f / 3168.0f) * k1x0 - (355.0f / 33.0f) * k2x0
                      + (46732.0f / 5247.0f) * k3x0 + (49.0f / 176.0f) * k4x0
                      - (5103.0f / 18656.0f) * k5x0),
           yy0 + h * ((9017.0f / 3168.0f) * k1y0 - (355.0f / 33.0f) * k2y0
                      + (46732.0f / 5247.0f) * k3y0 + (49.0f / 176.0f) * k4y0
                      - (5103.0f / 18656.0f) * k5y0),
           yx1 + h * ((9017.0f / 3168.0f) * k1x1 - (355.0f / 33.0f) * k2x1
                      + (46732.0f / 5247.0f) * k3x1 + (49.0f / 176.0f) * k4x1
                      - (5103.0f / 18656.0f) * k5x1),
           yy1 + h * ((9017.0f / 3168.0f) * k1y1 - (355.0f / 33.0f) * k2y1
                      + (46732.0f / 5247.0f) * k3y1 + (49.0f / 176.0f) * k4y1
                      - (5103.0f / 18656.0f) * k5y1),
           k6x0, k6y0, k6x1, k6y1);

    yx0 += h * ((35.0f / 384.0f) * k1x0 + (500.0f / 1113.0f) * k3x0 + (125.0f / 192.0f) * k4x0
                - (2187.0f / 6784.0f) * k5x0 + (11.0f / 84.0f) * k6x0);
    yy0 += h * ((35.0f / 384.0f) * k1y0 + (500.0f / 1113.0f) * k3y0 + (125.0f / 192.0f) * k4y0
                - (2187.0f / 6784.0f) * k5y0 + (11.0f / 84.0f) * k6y0);
    yx1 += h * ((35.0f / 384.0f) * k1x1 + (500.0f / 1113.0f) * k3x1 + (125.0f / 192.0f) * k4x1
                - (2187.0f / 6784.0f) * k5x1 + (11.0f / 84.0f) * k6x1);
    yy1 += h * ((35.0f / 384.0f) * k1y1 + (500.0f / 1113.0f) * k3y1 + (125.0f / 192.0f) * k4y1
                - (2187.0f / 6784.0f) * k5y1 + (11.0f / 84.0f) * k6y1);

    if (act0) outv[(size_t)(i + 1) * B + bb + m]      = make_float2(yx0, yy0);
    if (act1) outv[(size_t)(i + 1) * B + bb + 16 + m] = make_float2(yx1, yy1);
  }
}

extern "C" void kernel_launch(void* const* d_in, const int* in_sizes, int n_in,
                              void* d_out, int out_size, void* d_ws, size_t ws_size,
                              hipStream_t stream) {
  const float* y0 = (const float*)d_in[0];
  const float* t  = (const float*)d_in[1];
  const float* W1 = (const float*)d_in[2];
  const float* b1 = (const float*)d_in[3];
  const float* W2 = (const float*)d_in[4];
  const float* b2 = (const float*)d_in[5];
  float* out = (float*)d_out;
  const int B = in_sizes[0] / 2;   // 131072
  const int N = in_sizes[1];       // 128
  const int waves   = (B + 31) / 32;       // 32 batch elements (2 WMMA tiles) per wave32
  const int threads = 256;                 // 8 waves per block
  const int blocks  = (waves * 32 + threads - 1) / threads;
  node_dopri5_wmma<<<blocks, threads, 0, stream>>>(y0, t, W1, b1, W2, b2, out, B, N);
}
